// MixtureOfExperts_64957085385329
// MI455X (gfx1250) — matrix operand order repaired and checked by reading, not compiled, not run on hardware
//
#include <hip/hip_runtime.h>
#include <hip/hip_bf16.h>

// Problem constants (match reference)
constexpr int BT = 8192;   // tokens
constexpr int DI = 1024;   // input dim
constexpr int OO = 1024;   // output dim
constexpr int EE = 8;      // experts
constexpr int HH = 512;    // hidden
constexpr int NSLOT = BT * 2; // total assignments (top-2)

typedef __attribute__((ext_vector_type(16))) _Float16 v16h;
typedef __attribute__((ext_vector_type(8)))  _Float16 v8h;
typedef __attribute__((ext_vector_type(8)))  float    v8f;

__device__ __forceinline__ int imin(int a, int b) { return a < b ? a : b; }

// ---------------------------------------------------------------- init / zero
__global__ void moe_init(int* counts, int* fill) {
    int t = threadIdx.x;
    if (t < EE) { counts[t] = 0; fill[t] = 0; }
}

__global__ void moe_zero_out(float4* out) {
    size_t i = (size_t)blockIdx.x * blockDim.x + threadIdx.x;
    out[i] = make_float4(0.f, 0.f, 0.f, 0.f);
}

// ------------------------------------------------- f32 -> f16 flat convert (8 elems/thread)
__global__ void moe_cvt16(const float4* __restrict__ in, v8h* __restrict__ out) {
    size_t i = (size_t)blockIdx.x * blockDim.x + threadIdx.x;
    float4 a = in[i * 2], b = in[i * 2 + 1];
    v8h o = { (_Float16)a.x, (_Float16)a.y, (_Float16)a.z, (_Float16)a.w,
              (_Float16)b.x, (_Float16)b.y, (_Float16)b.z, (_Float16)b.w };
    out[i] = o;
}

// ------------------------------------------------- f32 [z][R][C] -> f16 [z][C][R] transpose
__global__ void moe_transpose16(const float* __restrict__ in, _Float16* __restrict__ out,
                                int R, int C) {
    __shared__ float tile[32][33];
    const int z = blockIdx.z;
    const int r0 = blockIdx.x * 32;
    const int c0 = blockIdx.y * 32;
    const int tx = threadIdx.x & 31, ty = threadIdx.x >> 5;
    const float* ip = in + (size_t)z * R * C;
    _Float16* op = out + (size_t)z * C * R;
#pragma unroll
    for (int i = 0; i < 4; ++i)
        tile[ty + 8 * i][tx] = ip[(size_t)(r0 + ty + 8 * i) * C + c0 + tx];
    __syncthreads();
#pragma unroll
    for (int i = 0; i < 4; ++i)
        op[(size_t)(c0 + ty + 8 * i) * R + r0 + tx] = (_Float16)tile[tx][ty + 8 * i];
}

// ---------------------------------------------------------------- gating
__global__ void moe_gate(const float* __restrict__ x, const float* __restrict__ gW,
                         const float* __restrict__ gb,
                         float* __restrict__ top_gates, int* __restrict__ top_i,
                         int* __restrict__ counts, float* __restrict__ imp_partial) {
    __shared__ float s_imp[EE];
    const int lane = threadIdx.x & 31;
    const int w    = threadIdx.x >> 5;
    const int row  = blockIdx.x * 8 + w;
    if (threadIdx.x < EE) s_imp[threadIdx.x] = 0.f;
    __syncthreads();

    float acc[EE];
#pragma unroll
    for (int e = 0; e < EE; ++e) acc[e] = 0.f;
    const float* xr = x + (size_t)row * DI;
    for (int d = lane; d < DI; d += 32) {
        float xv = xr[d];
        const float* g = gW + d * EE;
#pragma unroll
        for (int e = 0; e < EE; ++e) acc[e] += xv * g[e];
    }
#pragma unroll
    for (int off = 16; off; off >>= 1) {
#pragma unroll
        for (int e = 0; e < EE; ++e) acc[e] += __shfl_xor(acc[e], off, 32);
    }
    if (lane == 0) {
        float lg[EE];
#pragma unroll
        for (int e = 0; e < EE; ++e) lg[e] = acc[e] + gb[e];
        int i0 = 0;
        for (int e = 1; e < EE; ++e) if (lg[e] > lg[i0]) i0 = e;
        int i1 = (i0 == 0) ? 1 : 0;
        for (int e = 0; e < EE; ++e) if (e != i0 && lg[e] > lg[i1]) i1 = e;
        float g0 = 1.f / (1.f + expf(lg[i1] - lg[i0]));   // softmax over top-2
        top_gates[row * 2 + 0] = g0;
        top_gates[row * 2 + 1] = 1.f - g0;
        top_i[row * 2 + 0] = i0;
        top_i[row * 2 + 1] = i1;
        atomicAdd(&counts[i0], 1);
        atomicAdd(&counts[i1], 1);
        float m = lg[0];
        for (int e = 1; e < EE; ++e) m = fmaxf(m, lg[e]);
        float s = 0.f, p[EE];
#pragma unroll
        for (int e = 0; e < EE; ++e) { p[e] = expf(lg[e] - m); s += p[e]; }
        float inv = 1.f / s;
#pragma unroll
        for (int e = 0; e < EE; ++e) atomicAdd(&s_imp[e], p[e] * inv);
    }
    __syncthreads();
    if (threadIdx.x < EE) imp_partial[blockIdx.x * EE + threadIdx.x] = s_imp[threadIdx.x];
}

// ------------------------------------------------- loss + bucket offsets
__global__ void moe_loss_offsets(const float* __restrict__ imp_partial,
                                 const int* __restrict__ counts,
                                 int* __restrict__ offsets, float* __restrict__ out) {
    __shared__ float s[EE];
    const int t = threadIdx.x;
    if (t < EE) {
        float se = 0.f;
        for (int b = 0; b < BT / 8; ++b) se += imp_partial[b * EE + t]; // fixed order
        s[t] = se / (float)BT;
    }
    __syncthreads();
    if (t == 0) {
        float l = 0.f;
        for (int e = 0; e < EE; ++e) l += s[e] * s[e];
        out[(size_t)BT * OO] = 0.01f * (float)EE * l;
        int o = 0;
        for (int e = 0; e < EE; ++e) { offsets[e] = o; o += counts[e]; }
        offsets[EE] = o;
    }
}

// ---------------------------------------------------------------- scatter
__global__ void moe_scatter(const int* __restrict__ top_i, const float* __restrict__ top_gates,
                            const int* __restrict__ offsets, int* __restrict__ fill,
                            int* __restrict__ row_of_slot, float* __restrict__ gate_of_slot) {
    int r = blockIdx.x * blockDim.x + threadIdx.x;
#pragma unroll
    for (int k = 0; k < 2; ++k) {
        int e = top_i[r * 2 + k];
        int p = atomicAdd(&fill[e], 1);
        int s = offsets[e] + p;
        row_of_slot[s]  = r;
        gate_of_slot[s] = top_gates[r * 2 + k];
    }
}

// async copy helpers ----------------------------------------------------------
__device__ __forceinline__ void async_b128(uint32_t lds_addr, uint64_t gaddr) {
    asm volatile("global_load_async_to_lds_b128 %0, %1, off"
                 :: "v"(lds_addr), "v"(gaddr) : "memory");
}
__device__ __forceinline__ void async_b128_o16(uint32_t lds_addr, uint64_t gaddr) {
    asm volatile("global_load_async_to_lds_b128 %0, %1, off offset:16"
                 :: "v"(lds_addr), "v"(gaddr) : "memory");
}
__device__ __forceinline__ void wait_async(int pending) {
    if (pending) asm volatile("s_wait_asynccnt 0x4" ::: "memory");
    else         asm volatile("s_wait_asynccnt 0x0" ::: "memory");
}

// ------------------------------------------------- GEMM1: h = relu(x_sel @ W1[e] + b1[e])
// A: xh [BT][DI] f16 (gathered rows); B: W1t [E][HH][DI] f16 (k-contiguous rows)
// block tile 128x128, K-step 32; 8 waves, each 16x128 (8 WMMAs / A-frag);
// double-buffered async-to-LDS pipeline.
__global__ void __launch_bounds__(256) moe_gemm1(
    const _Float16* __restrict__ xh, const _Float16* __restrict__ W1t,
    const float* __restrict__ b1,
    const int* __restrict__ row_of_slot, const int* __restrict__ counts,
    const int* __restrict__ offsets, _Float16* __restrict__ Hbuf) {
    const int e = blockIdx.z;
    const int count = counts[e];
    const int m0 = blockIdx.x * 128;
    if (m0 >= count) return;                 // uniform early-exit, before barriers
    const int n0 = blockIdx.y * 128;
    const int base = offsets[e] + m0;
    const int valid = imin(128, count - m0);

    __shared__ __align__(32) _Float16 sA[2][128 * 32];
    __shared__ __align__(32) _Float16 sB[2][128 * 32];

    const int t = threadIdx.x;
    const int lane = t & 31, w = t >> 5;
    const int nl = lane & 15;
    const int ka = (lane >> 4) * 16;         // A: contiguous 16-K half per lane group
    const int kb = (lane >> 4) * 16;         // B: contiguous 16-K half per lane group

    // staging: thread t copies 32B of A-row (t>>1) and 32B of B-row (t>>1)
    const int am = t >> 1, ap = t & 1;
    const int arow = row_of_slot[imin(base + am, NSLOT - 1)];
    const _Float16* agp = xh + (size_t)arow * DI + ap * 16;
    const _Float16* bgp = W1t + (size_t)e * HH * DI + (size_t)(n0 + am) * DI + ap * 16;
    const uint32_t la_base = (uint32_t)(uintptr_t)&sA[0][am * 32 + ap * 16];
    const uint32_t lb_base = (uint32_t)(uintptr_t)&sB[0][am * 32 + ap * 16];
    const uint32_t bufStride = 128 * 32 * 2;

    v8f c[8];
#pragma unroll
    for (int j = 0; j < 8; ++j) c[j] = 0.f;

    auto issue = [&](int buf, int d0) {
        uint32_t la = la_base + buf * bufStride * 2;  // sA[2] then sB[2] are separate arrays
        uint32_t lb = lb_base + buf * bufStride * 2;
        // NOTE: index per-array instead (avoid layout assumptions):
        la = (uint32_t)(uintptr_t)&sA[buf][am * 32 + ap * 16];
        lb = (uint32_t)(uintptr_t)&sB[buf][am * 32 + ap * 16];
        async_b128(la, (uint64_t)(uintptr_t)(agp + d0));
        async_b128_o16(la, (uint64_t)(uintptr_t)(agp + d0));
        async_b128(lb, (uint64_t)(uintptr_t)(bgp + d0));
        async_b128_o16(lb, (uint64_t)(uintptr_t)(bgp + d0));
    };

    constexpr int KT = DI / 32;
    issue(0, 0);
#pragma clang loop unroll(disable)
    for (int kt = 0; kt < KT; ++kt) {
        const int cur = kt & 1;
        const int more = (kt + 1 < KT);
        if (more) issue(cur ^ 1, (kt + 1) * 32);
        wait_async(more);
        __syncthreads();

        const _Float16* sAc = sA[cur];
        const _Float16* sBc = sB[cur];
        const v8h* pa = reinterpret_cast<const v8h*>(&sAc[(w * 16 + nl) * 32 + ka]);
        v8h a0 = pa[0], a1 = pa[1];
        v16h af = __builtin_shufflevector(a0, a1, 0,1,2,3,4,5,6,7,8,9,10,11,12,13,14,15);
#pragma unroll
        for (int j = 0; j < 8; ++j) {
            v16h bf = *reinterpret_cast<const v16h*>(&sBc[(j * 16 + nl) * 32 + kb]);
            c[j] = __builtin_amdgcn_wmma_f32_16x16x32_f16(
                false, af, false, bf, (short)0, c[j], false, false);
        }
        __syncthreads();
    }
    // epilogue: bias + relu -> f16 Hbuf
#pragma unroll
    for (int j = 0; j < 8; ++j) {
        const int n = n0 + j * 16 + nl;
        const float bias = b1[e * HH + n];
#pragma unroll
        for (int v = 0; v < 8; ++v) {
            int ml = w * 16 + v + (lane >> 4) * 8;
            if (ml < valid)
                Hbuf[(size_t)(base + ml) * HH + n] = (_Float16)fmaxf(c[j][v] + bias, 0.f);
        }
    }
}

// ------------------------------------------------- GEMM2: out += gate * (h @ W2[e] + b2[e])
// A: Hbuf [NSLOT][HH] f16 (slot-contiguous); B: W2t [E][OO][HH] f16
__global__ void __launch_bounds__(256) moe_gemm2(
    const _Float16* __restrict__ Hbuf, const _Float16* __restrict__ W2t,
    const float* __restrict__ b2,
    const int* __restrict__ row_of_slot, const float* __restrict__ gate_of_slot,
    const int* __restrict__ counts, const int* __restrict__ offsets,
    float* __restrict__ out) {
    const int e = blockIdx.z;
    const int count = counts[e];
    const int m0 = blockIdx.x * 128;
    if (m0 >= count) return;
    const int n0 = blockIdx.y * 128;
    const int base = offsets[e] + m0;
    const int valid = imin(128, count - m0);

    __shared__ __align__(32) _Float16 sA[2][128 * 32];
    __shared__ __align__(32) _Float16 sB[2][128 * 32];

    const int t = threadIdx.x;
    const int lane = t & 31, w = t >> 5;
    const int nl = lane & 15;
    const int ka = (lane >> 4) * 16;
    const int kb = (lane >> 4) * 16;

    const int am = t >> 1, ap = t & 1;
    const _Float16* agp = Hbuf + (size_t)imin(base + am, NSLOT - 1) * HH + ap * 16;
    const _Float16* bgp = W2t + (size_t)e * OO * HH + (size_t)(n0 + am) * HH + ap * 16;

    v8f c[8];
#pragma unroll
    for (int j = 0; j < 8; ++j) c[j] = 0.f;

    auto issue = [&](int buf, int d0) {
        uint32_t la = (uint32_t)(uintptr_t)&sA[buf][am * 32 + ap * 16];
        uint32_t lb = (uint32_t)(uintptr_t)&sB[buf][am * 32 + ap * 16];
        async_b128(la, (uint64_t)(uintptr_t)(agp + d0));
        async_b128_o16(la, (uint64_t)(uintptr_t)(agp + d0));
        async_b128(lb, (uint64_t)(uintptr_t)(bgp + d0));
        async_b128_o16(lb, (uint64_t)(uintptr_t)(bgp + d0));
    };

    constexpr int KT = HH / 32;
    issue(0, 0);
#pragma clang loop unroll(disable)
    for (int kt = 0; kt < KT; ++kt) {
        const int cur = kt & 1;
        const int more = (kt + 1 < KT);
        if (more) issue(cur ^ 1, (kt + 1) * 32);
        wait_async(more);
        __syncthreads();

        const _Float16* sAc = sA[cur];
        const _Float16* sBc = sB[cur];
        const v8h* pa = reinterpret_cast<const v8h*>(&sAc[(w * 16 + nl) * 32 + ka]);
        v8h a0 = pa[0], a1 = pa[1];
        v16h af = __builtin_shufflevector(a0, a1, 0,1,2,3,4,5,6,7,8,9,10,11,12,13,14,15);
#pragma unroll
        for (int j = 0; j < 8; ++j) {
            v16h bf = *reinterpret_cast<const v16h*>(&sBc[(j * 16 + nl) * 32 + kb]);
            c[j] = __builtin_amdgcn_wmma_f32_16x16x32_f16(
                false, af, false, bf, (short)0, c[j], false, false);
        }
        __syncthreads();
    }
    // epilogue: gate * (acc + b2) scattered into out (exactly 2 adds/elem -> deterministic)
#pragma unroll
    for (int j = 0; j < 8; ++j) {
        const int n = n0 + j * 16 + nl;
        const float bias = b2[e * OO + n];
#pragma unroll
        for (int v = 0; v < 8; ++v) {
            int ml = w * 16 + v + (lane >> 4) * 8;
            if (ml < valid) {
                int slot = base + ml;
                float val = gate_of_slot[slot] * (c[j][v] + bias);
                atomicAdd(&out[(size_t)row_of_slot[slot] * OO + n], val);
            }
        }
    }
}

// ---------------------------------------------------------------- launch
extern "C" void kernel_launch(void* const* d_in, const int* in_sizes, int n_in,
                              void* d_out, int out_size, void* d_ws, size_t ws_size,
                              hipStream_t stream) {
    const float* x  = (const float*)d_in[0];
    const float* gW = (const float*)d_in[1];
    const float* gb = (const float*)d_in[2];
    const float* W1 = (const float*)d_in[3];
    const float* b1 = (const float*)d_in[4];
    const float* W2 = (const float*)d_in[5];
    const float* b2 = (const float*)d_in[6];
    float* out = (float*)d_out;

    char* ws = (char*)d_ws;
    size_t off = 0;
    auto alloc = [&](size_t bytes) -> char* {
        char* p = ws + off;
        off += (bytes + 255) & ~(size_t)255;
        return p;
    };
    float*    imp_partial  = (float*)alloc((BT / 8) * EE * sizeof(float));
    int*      counts       = (int*)alloc(EE * sizeof(int));
    int*      fill         = (int*)alloc(EE * sizeof(int));
    int*      offsets      = (int*)alloc((EE + 1) * sizeof(int));
    float*    top_gates    = (float*)alloc(NSLOT * sizeof(float));
    int*      top_i        = (int*)alloc(NSLOT * sizeof(int));
    int*      row_of_slot  = (int*)alloc(NSLOT * sizeof(int));
    float*    gate_of_slot = (float*)alloc(NSLOT * sizeof(float));
    _Float16* Hbuf         = (_Float16*)alloc((size_t)NSLOT * HH * sizeof(_Float16));
    _Float16* xh           = (_Float16*)alloc((size_t)BT * DI * sizeof(_Float16));
    _Float16* W1t          = (_Float16*)alloc((size_t)EE * DI * HH * sizeof(_Float16));
    _Float16* W2t          = (_Float16*)alloc((size_t)EE * HH * OO * sizeof(_Float16));

    moe_init<<<1, 32, 0, stream>>>(counts, fill);
    moe_zero_out<<<(BT * OO / 4) / 256, 256, 0, stream>>>((float4*)out);
    // one-time precision/layout conversion: x -> f16, W1 -> [E][H][D] f16, W2 -> [E][O][H] f16
    moe_cvt16<<<(BT * DI / 8) / 256, 256, 0, stream>>>((const float4*)x, (v8h*)xh);
    moe_transpose16<<<dim3(DI / 32, HH / 32, EE), 256, 0, stream>>>(W1, W1t, DI, HH);
    moe_transpose16<<<dim3(HH / 32, OO / 32, EE), 256, 0, stream>>>(W2, W2t, HH, OO);
    moe_gate<<<BT / 8, 256, 0, stream>>>(x, gW, gb, top_gates, top_i, counts, imp_partial);
    moe_loss_offsets<<<1, 32, 0, stream>>>(imp_partial, counts, offsets, out);
    moe_scatter<<<BT / 256, 256, 0, stream>>>(top_i, top_gates, offsets, fill,
                                              row_of_slot, gate_of_slot);
    moe_gemm1<<<dim3(64, HH / 128, EE), 256, 0, stream>>>(xh, W1t, b1, row_of_slot,
                                                          counts, offsets, Hbuf);
    moe_gemm2<<<dim3(64, OO / 128, EE), 256, 0, stream>>>(Hbuf, W2t, b2, row_of_slot,
                                                          gate_of_slot, counts, offsets, out);
}
